// MambaBlock_56487409877671
// MI455X (gfx1250) — matrix operand order repaired and checked by reading, not compiled, not run on hardware
//
#include <hip/hip_runtime.h>
#include <hip/hip_bf16.h>

typedef __bf16 bf16;
typedef __attribute__((ext_vector_type(16))) __bf16 v16bf;
typedef __attribute__((ext_vector_type(8)))  __bf16 v8bf;
typedef __attribute__((ext_vector_type(8)))  float  v8f;

// Problem constants
constexpr int B_  = 2;
constexpr int L_  = 2048;
constexpr int Dm_ = 512;
constexpr int E_  = 1024;
constexpr int N_  = 16;
constexpr int R_  = 32;
constexpr int M_  = B_ * L_;     // 4096 tokens

// ---------------------------------------------------------------------------
// WMMA fragment helpers (bf16 16x16x32, wave32)
// A (16x32, MxK): lanes 0-15 -> M=0..15 with K=0..15 contiguous; lanes 16-31 ->
//                 same M rows with K=16..31 contiguous. 16 bf16 per lane.
// B (32x16, KxN): lanes 0-15 -> N=0..15 with K=0..15 contiguous;
//                 lanes 16-31 -> N=0..15 with K=16..31 contiguous.
// ---------------------------------------------------------------------------
__device__ __forceinline__ v16bf load_a_frag(const bf16* __restrict__ X, int lda,
                                             int m0, int k0, int lane) {
  const int mrow = m0 + (lane & 15);
  const int kh   = (lane >> 4) << 4;          // 0 or 16
  const bf16* p  = X + (size_t)mrow * lda + k0;
  return *(const v16bf*)(p + kh);             // K = k0+kh .. +15 (32B load)
}

__device__ __forceinline__ v16bf load_b_frag(const bf16* __restrict__ W, int ldw,
                                             int n0, int k0, int lane) {
  const int nrow = n0 + (lane & 15);
  const int kh   = (lane >> 4) << 4;          // 0 or 16
  return *(const v16bf*)(W + (size_t)nrow * ldw + k0 + kh);  // 32B load
}

// One wave accumulates a 16(M) x 64(N) tile; A fragment reused across 4 WMMAs.
__device__ __forceinline__ void wave_tile_accum(const bf16* __restrict__ A_, int lda,
                                                const bf16* __restrict__ B_, int ldb,
                                                int m0, int n0, int Kdim, int lane,
                                                v8f acc[4]) {
  for (int k0 = 0; k0 < Kdim; k0 += 32) {
    v16bf a = load_a_frag(A_, lda, m0, k0, lane);
#pragma unroll
    for (int nt = 0; nt < 4; ++nt) {
      v16bf b = load_b_frag(B_, ldb, n0 + nt * 16, k0, lane);
      acc[nt] = __builtin_amdgcn_wmma_f32_16x16x32_bf16(
          false, a, false, b, (short)0, acc[nt], false, false);
    }
  }
}

// ---------------------------------------------------------------------------
// 1) RMSNorm -> bf16  (one 256-thread block per token row, Dm=512)
// ---------------------------------------------------------------------------
__global__ void rmsnorm_kernel(const float* __restrict__ x,
                               const float* __restrict__ w,
                               bf16* __restrict__ xn) {
  const int row = blockIdx.x;
  const float* xr = x + (size_t)row * Dm_;
  const int t = threadIdx.x;
  float a0 = xr[t], a1 = xr[t + 256];
  float ss = a0 * a0 + a1 * a1;
#pragma unroll
  for (int off = 16; off; off >>= 1) ss += __shfl_xor(ss, off, 32);
  __shared__ float red[8];
  if ((t & 31) == 0) red[t >> 5] = ss;
  __syncthreads();
  float tot = 0.f;
#pragma unroll
  for (int i = 0; i < 8; ++i) tot += red[i];
  const float scale = rsqrtf(tot * (1.0f / Dm_) + 1e-5f);
  bf16* o = xn + (size_t)row * Dm_;
  o[t]       = (bf16)(a0 * scale * w[t]);
  o[t + 256] = (bf16)(a1 * scale * w[t + 256]);
}

// ---------------------------------------------------------------------------
// 2) fp32 -> bf16 convert (weights)
// ---------------------------------------------------------------------------
__global__ void f32_to_bf16_kernel(const float* __restrict__ s,
                                   bf16* __restrict__ d, int n) {
  int i = blockIdx.x * blockDim.x + threadIdx.x;
  if (i < n) d[i] = (bf16)s[i];
}

// ---------------------------------------------------------------------------
// 3) in_proj: Xz[4096,2048] = Xn[4096,512] @ W[2048,512]^T  (bf16 out)
// ---------------------------------------------------------------------------
__global__ void gemm_in_proj_kernel(const bf16* __restrict__ Xn,
                                    const bf16* __restrict__ W,
                                    bf16* __restrict__ Xz) {
  const int wave = (blockIdx.x * blockDim.x + threadIdx.x) >> 5;
  const int lane = threadIdx.x & 31;
  const int n64 = wave & 31;          // 2048/64 = 32 N-tiles
  const int mt  = wave >> 5;          // 256 M-tiles
  const int m0 = mt * 16, n0 = n64 * 64;
  v8f acc[4] = {};
  wave_tile_accum(Xn, Dm_, W, Dm_, m0, n0, Dm_, lane, acc);
#pragma unroll
  for (int nt = 0; nt < 4; ++nt) {
    const int c = n0 + nt * 16 + (lane & 15);
#pragma unroll
    for (int i = 0; i < 8; ++i) {
      const int r = m0 + i + ((lane >> 4) << 3);
      Xz[(size_t)r * (2 * E_) + c] = (bf16)acc[nt][i];
    }
  }
}

// ---------------------------------------------------------------------------
// 4) causal depthwise conv (K=4) + SiLU -> xs (fp32 + bf16)
// ---------------------------------------------------------------------------
__global__ void conv_silu_kernel(const bf16* __restrict__ xz,
                                 const float* __restrict__ cw,
                                 const float* __restrict__ cb,
                                 float* __restrict__ xs,
                                 bf16* __restrict__ xsb) {
  const size_t i = (size_t)blockIdx.x * blockDim.x + threadIdx.x; // B*L*E
  const int e = (int)(i & (E_ - 1));
  const int m = (int)(i >> 10);
  const int l = m & (L_ - 1);
  const int b = m >> 11;
  float acc = cb[e];
#pragma unroll
  for (int k = 0; k < 4; ++k) {
    const int l2 = l - 3 + k;
    if (l2 >= 0)
      acc += cw[e * 4 + k] * (float)xz[((size_t)(b * L_ + l2)) * (2 * E_) + e];
  }
  const float s = acc / (1.f + __expf(-acc));
  xs[i] = s;
  xsb[i] = (bf16)s;
}

// ---------------------------------------------------------------------------
// 5) x_proj: dbc[4096,64] = xs[4096,1024] @ Wx[64,1024]^T ; dlt bf16 side-out
// ---------------------------------------------------------------------------
__global__ void gemm_xproj_kernel(const bf16* __restrict__ Xs,
                                  const bf16* __restrict__ Wx,
                                  float* __restrict__ dbc,
                                  bf16* __restrict__ dlt) {
  const int wave = (blockIdx.x * blockDim.x + threadIdx.x) >> 5;  // == m-tile
  const int lane = threadIdx.x & 31;
  const int m0 = wave * 16;
  v8f acc[4] = {};
  wave_tile_accum(Xs, E_, Wx, E_, m0, 0, E_, lane, acc);
#pragma unroll
  for (int nt = 0; nt < 4; ++nt) {
    const int c = nt * 16 + (lane & 15);
#pragma unroll
    for (int i = 0; i < 8; ++i) {
      const int r = m0 + i + ((lane >> 4) << 3);
      dbc[(size_t)r * 64 + c] = acc[nt][i];
      if (c < R_) dlt[(size_t)r * R_ + c] = (bf16)acc[nt][i];
    }
  }
}

// ---------------------------------------------------------------------------
// 6) dt_proj (K=32, single WMMA per tile) + bias + softplus -> delta fp32
// ---------------------------------------------------------------------------
__global__ void gemm_dt_kernel(const bf16* __restrict__ Dlt,
                               const bf16* __restrict__ Wdt,
                               const float* __restrict__ bias,
                               float* __restrict__ delta) {
  const int wave = (blockIdx.x * blockDim.x + threadIdx.x) >> 5;
  const int lane = threadIdx.x & 31;
  const int n64 = wave & 15;          // 1024/64 = 16
  const int mt  = wave >> 4;          // 256
  const int m0 = mt * 16, n0 = n64 * 64;
  v8f acc[4] = {};
  wave_tile_accum(Dlt, R_, Wdt, R_, m0, n0, R_, lane, acc);
#pragma unroll
  for (int nt = 0; nt < 4; ++nt) {
    const int c = n0 + nt * 16 + (lane & 15);
#pragma unroll
    for (int i = 0; i < 8; ++i) {
      const int r = m0 + i + ((lane >> 4) << 3);
      const float v = acc[nt][i] + bias[c];
      // numerically-stable softplus
      delta[(size_t)r * E_ + c] = fmaxf(v, 0.f) + log1pf(__expf(-fabsf(v)));
    }
  }
}

// ---------------------------------------------------------------------------
// 7) selective scan: block = 32 channels x 16 states; sequential over L.
//    thread(e_loc,n) owns h; y = sum_n h*C via 16-lane shfl_xor reduce.
// ---------------------------------------------------------------------------
__global__ void scan_kernel(const float* __restrict__ delta,
                            const float* __restrict__ xs,
                            const float* __restrict__ dbc,
                            const float* __restrict__ A,
                            float* __restrict__ ys) {
  const int tid = threadIdx.x;            // 512 threads
  const int n = tid & 15;
  const int e_loc = tid >> 4;             // 0..31
  const int eblk = blockIdx.x & 31;       // E/32 = 32
  const int b = blockIdx.x >> 5;
  const int e = eblk * 32 + e_loc;
  const float Aen = A[e * N_ + n];
  float h = 0.f;
  const size_t mbase = (size_t)b * L_;
  for (int l = 0; l < L_; ++l) {
    const size_t m = mbase + l;
    const float dl = delta[m * E_ + e];
    const float xv = xs[m * E_ + e];
    const float bm = dbc[m * 64 + R_ + n];
    const float cm = dbc[m * 64 + R_ + N_ + n];
    h = __expf(dl * Aen) * h + dl * bm * xv;
    float y = h * cm;
    y += __shfl_xor(y, 1, 32);
    y += __shfl_xor(y, 2, 32);
    y += __shfl_xor(y, 4, 32);
    y += __shfl_xor(y, 8, 32);
    if (n == 0) ys[m * E_ + e] = y;
  }
}

// ---------------------------------------------------------------------------
// 8) gating: yg = (ys + D*xs) * silu(z) -> bf16
// ---------------------------------------------------------------------------
__global__ void gate_kernel(const float* __restrict__ ys,
                            const float* __restrict__ xs,
                            const bf16* __restrict__ xz,
                            const float* __restrict__ Dv,
                            bf16* __restrict__ ygb) {
  const size_t i = (size_t)blockIdx.x * blockDim.x + threadIdx.x; // B*L*E
  const int e = (int)(i & (E_ - 1));
  const size_t m = i >> 10;
  const float zv = (float)xz[m * (2 * E_) + E_ + e];
  const float g = zv / (1.f + __expf(-zv));
  ygb[i] = (bf16)((ys[i] + Dv[e] * xs[i]) * g);
}

// ---------------------------------------------------------------------------
// 9) out_proj: out[4096,512] = yg[4096,1024] @ Wo[512,1024]^T + residual
// ---------------------------------------------------------------------------
__global__ void gemm_out_proj_kernel(const bf16* __restrict__ Yg,
                                     const bf16* __restrict__ Wo,
                                     const float* __restrict__ resid,
                                     float* __restrict__ out) {
  const int wave = (blockIdx.x * blockDim.x + threadIdx.x) >> 5;
  const int lane = threadIdx.x & 31;
  const int n64 = wave & 7;           // 512/64 = 8
  const int mt  = wave >> 3;          // 256
  const int m0 = mt * 16, n0 = n64 * 64;
  v8f acc[4] = {};
  wave_tile_accum(Yg, E_, Wo, E_, m0, n0, E_, lane, acc);
#pragma unroll
  for (int nt = 0; nt < 4; ++nt) {
    const int c = n0 + nt * 16 + (lane & 15);
#pragma unroll
    for (int i = 0; i < 8; ++i) {
      const int r = m0 + i + ((lane >> 4) << 3);
      out[(size_t)r * Dm_ + c] = acc[nt][i] + resid[(size_t)r * Dm_ + c];
    }
  }
}

// ---------------------------------------------------------------------------
extern "C" void kernel_launch(void* const* d_in, const int* in_sizes, int n_in,
                              void* d_out, int out_size, void* d_ws, size_t ws_size,
                              hipStream_t stream) {
  const float* x       = (const float*)d_in[0];
  const float* norm_w  = (const float*)d_in[1];
  const float* in_w    = (const float*)d_in[2];
  const float* conv_w  = (const float*)d_in[3];
  const float* conv_b  = (const float*)d_in[4];
  const float* xproj_w = (const float*)d_in[5];
  const float* dt_w    = (const float*)d_in[6];
  const float* dt_b    = (const float*)d_in[7];
  const float* Amat    = (const float*)d_in[8];
  const float* Dvec    = (const float*)d_in[9];
  const float* out_w   = (const float*)d_in[10];
  float* out = (float*)d_out;

  char* ws = (char*)d_ws;
  size_t off = 0;
  auto alloc = [&](size_t bytes) -> void* {
    void* p = ws + off;
    off += (bytes + 255) & ~(size_t)255;
    return p;
  };

  bf16*  xn_bf    = (bf16*)alloc((size_t)M_ * Dm_ * 2);       // 4 MB
  bf16*  in_w_bf  = (bf16*)alloc((size_t)2 * E_ * Dm_ * 2);   // 2 MB
  bf16*  xp_w_bf  = (bf16*)alloc((size_t)64 * E_ * 2);
  bf16*  dt_w_bf  = (bf16*)alloc((size_t)E_ * R_ * 2);
  bf16*  out_w_bf = (bf16*)alloc((size_t)Dm_ * E_ * 2);       // 1 MB
  bf16*  xz_bf    = (bf16*)alloc((size_t)M_ * 2 * E_ * 2);    // 16 MB
  float* xs_f     = (float*)alloc((size_t)M_ * E_ * 4);       // 16 MB
  bf16*  xs_bf    = (bf16*)alloc((size_t)M_ * E_ * 2);        // 8 MB
  float* dbc_f    = (float*)alloc((size_t)M_ * 64 * 4);       // 1 MB
  bf16*  dlt_bf   = (bf16*)alloc((size_t)M_ * R_ * 2);
  float* delta_f  = (float*)alloc((size_t)M_ * E_ * 4);       // 16 MB
  float* ys_f     = (float*)alloc((size_t)M_ * E_ * 4);       // 16 MB
  bf16*  ygb_bf   = (bf16*)alloc((size_t)M_ * E_ * 2);        // 8 MB

  // 1) RMSNorm
  rmsnorm_kernel<<<M_, 256, 0, stream>>>(x, norm_w, xn_bf);

  // 2) weight converts
  {
    int n;
    n = 2 * E_ * Dm_;
    f32_to_bf16_kernel<<<(n + 255) / 256, 256, 0, stream>>>(in_w, in_w_bf, n);
    n = 64 * E_;
    f32_to_bf16_kernel<<<(n + 255) / 256, 256, 0, stream>>>(xproj_w, xp_w_bf, n);
    n = E_ * R_;
    f32_to_bf16_kernel<<<(n + 255) / 256, 256, 0, stream>>>(dt_w, dt_w_bf, n);
    n = Dm_ * E_;
    f32_to_bf16_kernel<<<(n + 255) / 256, 256, 0, stream>>>(out_w, out_w_bf, n);
  }

  // 3) in_proj: 256 M-tiles * 32 N64-tiles = 8192 waves -> 1024 blocks
  gemm_in_proj_kernel<<<1024, 256, 0, stream>>>(xn_bf, in_w_bf, xz_bf);

  // 4) causal conv + SiLU over B*L*E = 4,194,304 elements
  conv_silu_kernel<<<(M_ * E_) / 256, 256, 0, stream>>>(xz_bf, conv_w, conv_b,
                                                        xs_f, xs_bf);

  // 5) x_proj: 256 waves -> 32 blocks
  gemm_xproj_kernel<<<32, 256, 0, stream>>>(xs_bf, xp_w_bf, dbc_f, dlt_bf);

  // 6) dt_proj + softplus: 256*16 = 4096 waves -> 512 blocks
  gemm_dt_kernel<<<512, 256, 0, stream>>>(dlt_bf, dt_w_bf, dt_b, delta_f);

  // 7) scan: B * (E/32) = 64 blocks of 512 threads
  scan_kernel<<<64, 512, 0, stream>>>(delta_f, xs_f, dbc_f, Amat, ys_f);

  // 8) gating
  gate_kernel<<<(M_ * E_) / 256, 256, 0, stream>>>(ys_f, xs_f, xz_bf, Dvec,
                                                   ygb_bf);

  // 9) out_proj + residual: 256*8 = 2048 waves -> 256 blocks
  gemm_out_proj_kernel<<<256, 256, 0, stream>>>(ygb_bf, out_w_bf, x, out);
}
